// HPNLearner_14147622273651
// MI455X (gfx1250) — matrix operand order repaired and checked by reading, not compiled, not run on hardware
//
#include <hip/hip_runtime.h>
#include <hip/hip_bf16.h>

typedef __attribute__((ext_vector_type(16))) __bf16 v16bf;
typedef __attribute__((ext_vector_type(8)))  float  v8f;

#define SS   13
#define SS2  169
#define SS4  28561
#define NGRP 4
#define KTMAX 1152   // max K (=128*9), also max table size

__device__ __forceinline__ unsigned short f2bf_rne(float f) {
  unsigned int u = __float_as_uint(f);
  unsigned int r = 0x7fffu + ((u >> 16) & 1u);
  return (unsigned short)((u + r) >> 16);
}

// ---------------- f32 -> raw bf16 ----------------
__global__ void f32_to_bf16_kernel(const float* __restrict__ in,
                                   unsigned short* __restrict__ out, int n) {
  for (int i = blockIdx.x * blockDim.x + threadIdx.x; i < n;
       i += gridDim.x * blockDim.x)
    out[i] = f2bf_rne(in[i]);
}

// ---------------- pack conv weights into WMMA A-fragment order ----------------
// apack element index e = ((mt*KB + kb)*32 + lane)*16 + j
// k order used here (B panel matches): lanes 0-15 -> K {0..15}; lanes 16-31 -> K {16..31}
__global__ void pack_weights_kernel(const float* __restrict__ w, // [Cout,Cin,3,3]
                                    unsigned short* __restrict__ apack,
                                    int Cin, int K, int KB, int total) {
  for (int e = blockIdx.x * blockDim.x + threadIdx.x; e < total;
       e += gridDim.x * blockDim.x) {
    int j    = e & 15;
    int lane = (e >> 4) & 31;
    int rest = e >> 9;
    int kb   = rest % KB;
    int mt   = rest / KB;
    int half = (lane < 16) ? 0 : 16;
    int kl   = half + j;
    int k    = kb * 32 + kl;
    int m    = mt * 16 + (lane & 15);
    unsigned short v = 0;
    if (k < K) {
      int cin = k / 9, rr = k % 9;
      int kh = rr / 3, kw = rr % 3;
      v = f2bf_rne(w[((m * Cin + cin) * 3 + kh) * 3 + kw]);
    }
    apack[e] = v;
  }
}

// ---------------- CenterPivotConv4d as implicit GEMM on WMMA ----------------
// grid: (ceil(169/16), B*169 (b,ha,wa), Cout/64); block: 128 threads = 4 waves.
// Per-block LDS tables hold resolved gather offsets; the hot loop does
// branch-free speculative gathers with unsigned 32-bit offsets (saddr+voffset
// addressing) and a double-buffered B panel (one barrier per k-step).
__global__ void __launch_bounds__(128)
conv4d_wmma_kernel(const unsigned short* __restrict__ act,    // bf16 [B,Cin,13,13,13,13]
                   const unsigned short* __restrict__ apack1, // conv over (ha,wa)
                   const unsigned short* __restrict__ apack2, // conv over (hb,wb)
                   const float* __restrict__ bias1,
                   const float* __restrict__ bias2,
                   float* __restrict__ y,                     // f32 [B,Cout,13,13,13,13]
                   int Cin, int Cout, int K, int KB) {
  __shared__ __align__(32) unsigned short Bpanel[2][16 * 32]; // [buf][n][k] bf16
  __shared__ int off1s[KTMAX];
  __shared__ int off2s[KTMAX];
  __shared__ signed char d1s[KTMAX];
  __shared__ signed char d2s[KTMAX];

  const int tid  = threadIdx.x;
  const int lane = tid & 31;
  const int wave = tid >> 5;
  const int qt   = blockIdx.x;          // position tile in (hb,wb) plane
  const int pe   = blockIdx.y;          // b*169 + ha*13 + wa
  const int mblk = blockIdx.z;          // 64-channel block
  const int b    = pe / SS2;
  const int rha  = pe % SS2;
  const int ha   = rha / SS, wa = rha % SS;
  const int p0   = qt * 16;
  const int mt   = mblk * 4 + wave;     // 16-row A tile index
  const int coutBase = mt * 16;
  const int KT   = KB * 32;

  // ---- build per-block gather tables (amortized over all k-steps) ----
  for (int kg = tid; kg < KT; kg += 128) {
    int o1 = (int)0x80000000;           // sentinel: tap out of range / k >= K
    int o2 = 0;
    signed char dh8 = 127, dw8 = 127;   // sentinel fails the bounds check
    if (kg < K) {
      int cin = kg / 9, rr = kg % 9;
      int dh = rr / 3 - 1, dw = rr % 3 - 1;
      int cb = b * Cin + cin;
      int ia = ha + dh, ja = wa + dw;
      if ((unsigned)ia < (unsigned)SS && (unsigned)ja < (unsigned)SS)
        o1 = ((cb * SS + ia) * SS + ja) * SS2;
      o2 = ((cb * SS + ha) * SS + wa) * SS2 + dh * SS + dw;
      dh8 = (signed char)dh;
      dw8 = (signed char)dw;
    }
    off1s[kg] = o1;
    off2s[kg] = o2;
    d1s[kg]   = dh8;
    d2s[kg]   = dw8;
  }
  __syncthreads();

  // per-thread staging constants: column n is invariant across the 4 slots
  const int n   = tid & 15;
  const int k0  = tid >> 4;             // 0..7
  const int q   = p0 + n;
  const bool qv = q < SS2;
  const int hb  = q / SS;
  const int wb  = q - hb * SS;

  v8f acc = {};
  int it = 0;                           // k-step counter across both convs

  for (int conv = 0; conv < 2; ++conv) {
    const unsigned short* ap =
        (conv == 0 ? apack1 : apack2) + (size_t)mt * KB * 512;
    for (int kb = 0; kb < KB; ++kb, ++it) {
      // A fragment: independent of LDS, issue early
      v16bf afrag = *(const v16bf*)(ap + ((size_t)kb * 32 + lane) * 16);
      if (kb + 1 < KB)
        __builtin_prefetch(ap + ((size_t)(kb + 1) * 32 + lane) * 16, 0, 3);

      // branch-free speculative gathers into registers
      unsigned int gv[4];
#pragma unroll
      for (int i = 0; i < 4; ++i) {
        int k  = k0 + i * 8;            // 0..31
        int kg = kb * 32 + k;
        unsigned int addr;
        bool ok;
        if (conv == 0) {
          int o = off1s[kg];
          ok   = qv && (o >= 0);
          addr = ok ? (unsigned int)(o + q) : 0u;
        } else {
          int dh = d1s[kg], dw = d2s[kg];
          ok   = qv && (unsigned)(hb + dh) < (unsigned)SS &&
                       (unsigned)(wb + dw) < (unsigned)SS;
          addr = ok ? (unsigned int)(off2s[kg] + q) : 0u;
        }
        unsigned int t = act[addr];     // unconditional load (addr always valid)
        gv[i] = ok ? t : 0u;
      }

      unsigned short* panel = Bpanel[it & 1];
#pragma unroll
      for (int i = 0; i < 4; ++i)
        panel[n * 32 + k0 + i * 8] = (unsigned short)gv[i];
      __syncthreads();                  // writes visible; safe vs stragglers (2 bufs)

      v16bf bfrag =
          *(const v16bf*)(&panel[(lane & 15) * 32 + ((lane < 16) ? 0 : 16)]);
      acc = __builtin_amdgcn_wmma_f32_16x16x32_bf16(
          false, afrag, false, bfrag, (short)0, acc, false, false);
    }
  }

  // D layout: VGPR v, lane l -> M = v + 8*(l>=16), N = l&15
  const int nq    = p0 + (lane & 15);
  const int mhalf = (lane >> 4) * 8;
  if (nq < SS2) {
#pragma unroll
    for (int v = 0; v < 8; ++v) {
      int cout = coutBase + mhalf + v;
      float val = acc[v] + bias1[cout] + bias2[cout];
      y[((size_t)(b * Cout + cout) * SS2 + rha) * SS2 + nq] = val;
    }
  }
}

// ---------------- GroupNorm statistics: one block per (b, group) ----------------
__global__ void __launch_bounds__(256)
gn_stats_kernel(const float* __restrict__ y, float* __restrict__ stats, int Cout) {
  const int bg = blockIdx.x;            // b*4 + g
  const int b  = bg >> 2, g = bg & 3;
  const int cg = Cout / NGRP;
  const size_t n = (size_t)cg * SS4;
  const float* base = y + ((size_t)b * Cout + (size_t)g * cg) * SS4;
  float s = 0.f, s2 = 0.f;
  for (size_t i = threadIdx.x; i < n; i += 256) {
    float v = base[i];
    s += v;
    s2 += v * v;
  }
  __shared__ float sh[256], sh2[256];
  sh[threadIdx.x] = s;
  sh2[threadIdx.x] = s2;
  __syncthreads();
  for (int st = 128; st > 0; st >>= 1) {
    if (threadIdx.x < st) {
      sh[threadIdx.x] += sh[threadIdx.x + st];
      sh2[threadIdx.x] += sh2[threadIdx.x + st];
    }
    __syncthreads();
  }
  if (threadIdx.x == 0) {
    float inv = 1.0f / (float)n;
    float mean = sh[0] * inv;
    float var = sh2[0] * inv - mean * mean;
    stats[bg * 2]     = mean;
    stats[bg * 2 + 1] = rsqrtf(var + 1e-5f);
  }
}

// ---------------- GN + ReLU -> bf16 activation (layers 0,1) ----------------
__global__ void gn_relu_bf16_kernel(const float* __restrict__ y,
                                    const float* __restrict__ stats,
                                    const float* __restrict__ gs,
                                    const float* __restrict__ gb,
                                    unsigned short* __restrict__ act,
                                    int Cout, int total) {
  const int cg = Cout / NGRP;
  for (int i = blockIdx.x * blockDim.x + threadIdx.x; i < total;
       i += gridDim.x * blockDim.x) {
    int c = (i / SS4) % Cout;
    int b = i / (SS4 * Cout);
    int g = c / cg;
    float mean = stats[(b * 4 + g) * 2];
    float rstd = stats[(b * 4 + g) * 2 + 1];
    float v = (y[i] - mean) * rstd * gs[c] + gb[c];
    v = v > 0.f ? v : 0.f;
    act[i] = f2bf_rne(v);
  }
}

// ---------------- final: GN + ReLU + mean over wb (layer 2) ----------------
__global__ void gn_relu_mean_kernel(const float* __restrict__ y,
                                    const float* __restrict__ stats,
                                    const float* __restrict__ gs,
                                    const float* __restrict__ gb,
                                    float* __restrict__ out, int total) {
  // out index i enumerates (b, c(256), ha, wa, hb); y row = y + i*13
  for (int i = blockIdx.x * blockDim.x + threadIdx.x; i < total;
       i += gridDim.x * blockDim.x) {
    int c = (i / (SS * SS * SS)) % 256;
    int b = i / (256 * SS * SS * SS);
    int g = c >> 6;                     // 256/4 = 64 channels per group
    float mean = stats[(b * 4 + g) * 2];
    float rstd = stats[(b * 4 + g) * 2 + 1];
    float sc = rstd * gs[c];
    float sh = gb[c] - mean * sc;
    const float* base = y + (size_t)i * SS;
    float s = 0.f;
#pragma unroll
    for (int wb = 0; wb < SS; ++wb) {
      float v = base[wb] * sc + sh;
      s += v > 0.f ? v : 0.f;
    }
    out[i] = s * (1.0f / (float)SS);
  }
}

// ================================================================ host side
static inline int gsblocks(long long n, int thr) {
  long long b = (n + thr - 1) / thr;
  if (b > 16384) b = 16384;
  return (int)b;
}

extern "C" void kernel_launch(void* const* d_in, const int* in_sizes, int n_in,
                              void* d_out, int out_size, void* d_ws, size_t ws_size,
                              hipStream_t stream) {
  (void)in_sizes; (void)n_in; (void)ws_size;
  const float* x = (const float*)d_in[0];
  const float* w1[3]; const float* b1[3]; const float* w2[3]; const float* b2[3];
  const float* gs[3]; const float* gb[3];
  for (int i = 0; i < 3; ++i) {
    w1[i] = (const float*)d_in[1 + 6 * i + 0];
    b1[i] = (const float*)d_in[1 + 6 * i + 1];
    w2[i] = (const float*)d_in[1 + 6 * i + 2];
    b2[i] = (const float*)d_in[1 + 6 * i + 3];
    gs[i] = (const float*)d_in[1 + 6 * i + 4];
    gb[i] = (const float*)d_in[1 + 6 * i + 5];
  }

  const int B = 4;
  const int chans[4] = {16, 64, 128, 256};
  const int Kk[3]  = {16 * 9, 64 * 9, 128 * 9};
  const int KBk[3] = {(16 * 9 + 31) / 32, (64 * 9) / 32, (128 * 9) / 32}; // 5,18,36

  size_t off = 0;
  auto carve = [&](size_t bytes) {
    size_t o = off;
    off += (bytes + 255) & ~(size_t)255;
    return o;
  };
  char* ws = (char*)d_ws;
  const size_t n_x    = (size_t)B * 16 * SS4;
  const size_t n_act0 = (size_t)B * 64 * SS4;
  const size_t n_act1 = (size_t)B * 128 * SS4;
  const size_t n_ymax = (size_t)B * 256 * SS4;

  unsigned short* xbf  = (unsigned short*)(ws + carve(n_x * 2));
  unsigned short* act0 = (unsigned short*)(ws + carve(n_act0 * 2));
  unsigned short* act1 = (unsigned short*)(ws + carve(n_act1 * 2));
  float*          ybuf = (float*)(ws + carve(n_ymax * 4));
  unsigned short* ap1[3]; unsigned short* ap2[3];
  int apTotal[3];
  for (int i = 0; i < 3; ++i) {
    int MT = chans[i + 1] / 16;
    apTotal[i] = MT * KBk[i] * 512;               // elements per conv
    ap1[i] = (unsigned short*)(ws + carve((size_t)apTotal[i] * 2));
    ap2[i] = (unsigned short*)(ws + carve((size_t)apTotal[i] * 2));
  }
  float* stats = (float*)(ws + carve(32 * sizeof(float)));

  // 1) x -> bf16
  f32_to_bf16_kernel<<<gsblocks(n_x, 256), 256, 0, stream>>>(x, xbf, (int)n_x);

  // 2) pack all weights into WMMA A-fragment order
  for (int i = 0; i < 3; ++i) {
    pack_weights_kernel<<<gsblocks(apTotal[i], 256), 256, 0, stream>>>(
        w1[i], ap1[i], chans[i], Kk[i], KBk[i], apTotal[i]);
    pack_weights_kernel<<<gsblocks(apTotal[i], 256), 256, 0, stream>>>(
        w2[i], ap2[i], chans[i], Kk[i], KBk[i], apTotal[i]);
  }

  const unsigned short* actIn[3] = {xbf, act0, act1};
  unsigned short* actOut[2] = {act0, act1};
  const int qtiles = (SS2 + 15) / 16; // 11

  for (int L = 0; L < 3; ++L) {
    int Cin = chans[L], Cout = chans[L + 1];
    dim3 grid(qtiles, B * SS2, Cout / 64);
    conv4d_wmma_kernel<<<grid, 128, 0, stream>>>(
        actIn[L], ap1[L], ap2[L], b1[L], b2[L], ybuf, Cin, Cout, Kk[L], KBk[L]);
    gn_stats_kernel<<<B * NGRP, 256, 0, stream>>>(ybuf, stats, Cout);
    if (L < 2) {
      int total = B * Cout * SS4;
      gn_relu_bf16_kernel<<<gsblocks(total, 256), 256, 0, stream>>>(
          ybuf, stats, gs[L], gb[L], actOut[L], Cout, total);
    } else {
      gn_relu_mean_kernel<<<gsblocks(out_size, 256), 256, 0, stream>>>(
          ybuf, stats, gs[L], gb[L], (float*)d_out, out_size);
    }
  }
}
